// GNN_22823456211680
// MI455X (gfx1250) — matrix operand order repaired and checked by reading, not compiled, not run on hardware
//
#include <hip/hip_runtime.h>
#include <hip/hip_bf16.h>

#define N_NODES 50000
#define N_EDGES 640000
#define HID 128
#define WAVES 4

typedef __attribute__((ext_vector_type(16))) __bf16 v16bf;
typedef __attribute__((ext_vector_type(8)))  float  v8f;

union FragU { uint4 u[2]; v16bf v; };

// A fragment for v_wmma_f32_16x16x32_bf16 from a row-major bf16 row in LDS.
// ISA 16-bit A 16x32 layout: lanes 0-15: v0..3=K0..7, v4..7=K16..23;
//                            lanes16-31: v0..3=K8..15, v4..7=K24..31.
__device__ __forceinline__ v16bf load_a_frag(const __bf16* rowp, int kt, int hf) {
  FragU t;
  const __bf16* p = rowp + kt * 32 + 8 * hf;
  t.u[0] = *(const uint4*)(p);
  t.u[1] = *(const uint4*)(p + 16);
  return t.v;
}

// B fragment: pre-packed so each lane's 16 values are contiguous (32B).
__device__ __forceinline__ v16bf load_b_frag(const __bf16* p) {
  FragU t;
  const uint4* q = (const uint4*)p;
  t.u[0] = q[0];
  t.u[1] = q[1];
  return t.v;
}

__device__ __forceinline__ v8f wmma_bf16(v16bf a, v16bf b, v8f c) {
  return __builtin_amdgcn_wmma_f32_16x16x32_bf16(false, a, false, b, (short)0, c,
                                                 false, false);
}

__device__ __forceinline__ float silu_f(float x) {
  return x / (1.0f + __expf(-x));
}

// ---------------------------------------------------------------------------
// Pack a K x 128 f32 matrix (row stride = ld) into WMMA B-fragment bf16 layout:
// dst[((nt*(K/32)+kt)*32 + lane)*16 + j] = src[(kt*32 + 8*(lane>>4) + (j&7) + 16*(j>>3))*ld + nt*16 + (lane&15)]
// ---------------------------------------------------------------------------
__global__ void pack_b_kernel(const float* __restrict__ src, __bf16* __restrict__ dst,
                              int K, int ld) {
  int idx = blockIdx.x * 256 + threadIdx.x;
  int total = K * 128;
  if (idx >= total) return;
  int j    = idx & 15;
  int lane = (idx >> 4) & 31;
  int t    = idx >> 9;
  int KT   = K >> 5;
  int nt   = t / KT;
  int kt   = t - nt * KT;
  int k    = kt * 32 + ((lane >> 4) << 3) + (j & 7) + ((j >> 3) << 4);
  int n    = nt * 16 + (lane & 15);
  dst[idx] = (__bf16)src[(size_t)k * ld + n];
}

__global__ void zero_f4_kernel(float4* __restrict__ p, long n4) {
  long i = (long)blockIdx.x * 256 + threadIdx.x;
  if (i < n4) p[i] = make_float4(0.f, 0.f, 0.f, 0.f);
}

// h[n][j] = nodes[n] @ emb_w + emb_b   (K=3, pure VALU, bandwidth bound)
__global__ void embed_kernel(const float* __restrict__ nodes, const float* __restrict__ w,
                             const float* __restrict__ b, __bf16* __restrict__ h) {
  long i = (long)blockIdx.x * 256 + threadIdx.x;
  if (i >= (long)N_NODES * HID) return;
  long n = i >> 7;
  int  j = (int)(i & 127);
  float acc = b[j] + nodes[n * 3 + 0] * w[0 * HID + j]
                   + nodes[n * 3 + 1] * w[1 * HID + j]
                   + nodes[n * 3 + 2] * w[2 * HID + j];
  h[i] = (__bf16)acc;
}

// ---------------------------------------------------------------------------
// Edge kernel: per wave, 16 edges. m = silu(silu([h_r|h_c|attr]W1+b1)W2+b2),
// scatter-add into agg[row].
// ---------------------------------------------------------------------------
__global__ __launch_bounds__(WAVES * 32) void edge_kernel(
    const __bf16* __restrict__ h, const int* __restrict__ row,
    const int* __restrict__ col, const float* __restrict__ eattr,
    const __bf16* __restrict__ w1p, const float* __restrict__ w1last,
    const float* __restrict__ b1, const __bf16* __restrict__ w2p,
    const float* __restrict__ b2, float* __restrict__ agg) {
  __shared__ __bf16 sA[WAVES][16][264];  // [h_src | h_tgt], row-major over K=256
  __shared__ __bf16 sM[WAVES][16][136];  // intermediate messages

  int tid  = threadIdx.x;
  int wave = tid >> 5;
  int lane = tid & 31;
  int hf   = lane >> 4;
  int lm   = lane & 15;
  long ebase = ((long)blockIdx.x * WAVES + wave) * 16;

  // Stage A: two lanes per edge; lane<16 gathers h[row], lane>=16 gathers h[col]
  {
    long e = ebase + lm;
    int src = hf ? col[e] : row[e];
    const uint4* gp = (const uint4*)(h + (size_t)src * HID);
    uint4* lp = (uint4*)(&sA[wave][lm][hf * 128]);
#pragma unroll
    for (int j = 0; j < 16; ++j) lp[j] = gp[j];
  }
  __syncthreads();

  float att[8];
  int   rw[8];
#pragma unroll
  for (int r = 0; r < 8; ++r) {
    long e = ebase + r + 8 * hf;
    att[r] = eattr[e];
    rw[r]  = row[e];
  }

  // GEMM1: 16x128 = (16x256)(256x128), K-loop of 8 WMMAs per N-tile
  for (int nt = 0; nt < 8; ++nt) {
    v8f acc = {};
#pragma unroll
    for (int kt = 0; kt < 8; ++kt) {
      v16bf a = load_a_frag(&sA[wave][lm][0], kt, hf);
      v16bf b = load_b_frag(w1p + ((size_t)(nt * 8 + kt) * 32 + lane) * 16);
      acc = wmma_bf16(a, b, acc);
    }
    int n = nt * 16 + lm;
    float wl = w1last[n];
    float bb = b1[n];
#pragma unroll
    for (int r = 0; r < 8; ++r) {
      int m = r + 8 * hf;
      float x = acc[r] + att[r] * wl + bb;
      sM[wave][m][n] = (__bf16)silu_f(x);
    }
  }
  __syncthreads();

  // GEMM2: 16x128 = (16x128)(128x128), then scatter-add
  for (int nt = 0; nt < 8; ++nt) {
    v8f acc = {};
#pragma unroll
    for (int kt = 0; kt < 4; ++kt) {
      v16bf a = load_a_frag(&sM[wave][lm][0], kt, hf);
      v16bf b = load_b_frag(w2p + ((size_t)(nt * 4 + kt) * 32 + lane) * 16);
      acc = wmma_bf16(a, b, acc);
    }
    int n = nt * 16 + lm;
    float bb = b2[n];
#pragma unroll
    for (int r = 0; r < 8; ++r) {
      float msg = silu_f(acc[r] + bb);
      atomicAdd(&agg[(size_t)rw[r] * HID + n], msg);
    }
  }
}

// ---------------------------------------------------------------------------
// Node kernel: h_new = silu([h|agg]W1+b1)W2+b2
// ---------------------------------------------------------------------------
__global__ __launch_bounds__(WAVES * 32) void node_kernel(
    const __bf16* __restrict__ h, const float* __restrict__ agg,
    const __bf16* __restrict__ w1p, const float* __restrict__ b1,
    const __bf16* __restrict__ w2p, const float* __restrict__ b2,
    __bf16* __restrict__ hout) {
  __shared__ __bf16 sA[WAVES][16][264];
  __shared__ __bf16 sM[WAVES][16][136];

  int tid  = threadIdx.x;
  int wave = tid >> 5;
  int lane = tid & 31;
  int hf   = lane >> 4;
  int lm   = lane & 15;
  long nbase = ((long)blockIdx.x * WAVES + wave) * 16;

  {
    long node = nbase + lm;
    if (node >= N_NODES) node = N_NODES - 1;  // clamp (stores are guarded)
    if (!hf) {
      const uint4* gp = (const uint4*)(h + (size_t)node * HID);
      uint4* lp = (uint4*)(&sA[wave][lm][0]);
#pragma unroll
      for (int j = 0; j < 16; ++j) lp[j] = gp[j];
    } else {
      const float4* gp = (const float4*)(agg + (size_t)node * HID);
      __bf16* lp = &sA[wave][lm][128];
#pragma unroll 8
      for (int j = 0; j < 32; ++j) {
        float4 f = gp[j];
        lp[4 * j + 0] = (__bf16)f.x;
        lp[4 * j + 1] = (__bf16)f.y;
        lp[4 * j + 2] = (__bf16)f.z;
        lp[4 * j + 3] = (__bf16)f.w;
      }
    }
  }
  __syncthreads();

  for (int nt = 0; nt < 8; ++nt) {
    v8f acc = {};
#pragma unroll
    for (int kt = 0; kt < 8; ++kt) {
      v16bf a = load_a_frag(&sA[wave][lm][0], kt, hf);
      v16bf b = load_b_frag(w1p + ((size_t)(nt * 8 + kt) * 32 + lane) * 16);
      acc = wmma_bf16(a, b, acc);
    }
    int n = nt * 16 + lm;
    float bb = b1[n];
#pragma unroll
    for (int r = 0; r < 8; ++r) {
      int m = r + 8 * hf;
      sM[wave][m][n] = (__bf16)silu_f(acc[r] + bb);
    }
  }
  __syncthreads();

  for (int nt = 0; nt < 8; ++nt) {
    v8f acc = {};
#pragma unroll
    for (int kt = 0; kt < 4; ++kt) {
      v16bf a = load_a_frag(&sM[wave][lm][0], kt, hf);
      v16bf b = load_b_frag(w2p + ((size_t)(nt * 4 + kt) * 32 + lane) * 16);
      acc = wmma_bf16(a, b, acc);
    }
    int n = nt * 16 + lm;
    float bb = b2[n];
#pragma unroll
    for (int r = 0; r < 8; ++r) {
      long node = nbase + r + 8 * hf;
      if (node < N_NODES) hout[(size_t)node * HID + n] = (__bf16)(acc[r] + bb);
    }
  }
}

// ---------------------------------------------------------------------------
// Decoder: out = silu(h W1 + b1) W2 + b2   (W2 is 128x3 -> VALU tail)
// ---------------------------------------------------------------------------
__global__ __launch_bounds__(WAVES * 32) void dec_kernel(
    const __bf16* __restrict__ h, const __bf16* __restrict__ w1p,
    const float* __restrict__ b1, const float* __restrict__ w2,
    const float* __restrict__ b2, float* __restrict__ out) {
  __shared__ __bf16 sA[WAVES][16][136];
  __shared__ float  sT[WAVES][16][132];

  int tid  = threadIdx.x;
  int wave = tid >> 5;
  int lane = tid & 31;
  int hf   = lane >> 4;
  int lm   = lane & 15;
  long nbase = ((long)blockIdx.x * WAVES + wave) * 16;

  {
    long node = nbase + lm;
    if (node >= N_NODES) node = N_NODES - 1;
    const uint4* gp = (const uint4*)(h + (size_t)node * HID) + hf * 8;
    uint4* lp = (uint4*)(&sA[wave][lm][hf * 64]);
#pragma unroll
    for (int j = 0; j < 8; ++j) lp[j] = gp[j];
  }
  __syncthreads();

  for (int nt = 0; nt < 8; ++nt) {
    v8f acc = {};
#pragma unroll
    for (int kt = 0; kt < 4; ++kt) {
      v16bf a = load_a_frag(&sA[wave][lm][0], kt, hf);
      v16bf b = load_b_frag(w1p + ((size_t)(nt * 4 + kt) * 32 + lane) * 16);
      acc = wmma_bf16(a, b, acc);
    }
    int n = nt * 16 + lm;
    float bb = b1[n];
#pragma unroll
    for (int r = 0; r < 8; ++r) {
      int m = r + 8 * hf;
      sT[wave][m][n] = silu_f(acc[r] + bb);
    }
  }
  __syncthreads();

  for (int it = 0; it < 2; ++it) {
    int id = lane + 32 * it;
    if (id < 48) {
      int m = id / 3, c = id - 3 * m;
      long node = nbase + m;
      if (node < N_NODES) {
        float s = b2[c];
        for (int k = 0; k < HID; ++k) s += sT[wave][m][k] * w2[k * 3 + c];
        out[node * 3 + c] = s;
      }
    }
  }
}

// ---------------------------------------------------------------------------
extern "C" void kernel_launch(void* const* d_in, const int* in_sizes, int n_in,
                              void* d_out, int out_size, void* d_ws, size_t ws_size,
                              hipStream_t stream) {
  (void)in_sizes; (void)n_in; (void)out_size; (void)ws_size;
  const float* nodes   = (const float*)d_in[0];
  const int*   edges   = (const int*)d_in[1];
  const float* eattr   = (const float*)d_in[2];
  const float* emb_w   = (const float*)d_in[3];
  const float* emb_b   = (const float*)d_in[4];
  const float* edge_w1 = (const float*)d_in[5];
  const float* edge_b1 = (const float*)d_in[6];
  const float* edge_w2 = (const float*)d_in[7];
  const float* edge_b2 = (const float*)d_in[8];
  const float* node_w1 = (const float*)d_in[9];
  const float* node_b1 = (const float*)d_in[10];
  const float* node_w2 = (const float*)d_in[11];
  const float* node_b2 = (const float*)d_in[12];
  const float* dec_w1  = (const float*)d_in[13];
  const float* dec_b1  = (const float*)d_in[14];
  const float* dec_w2  = (const float*)d_in[15];
  const float* dec_b2  = (const float*)d_in[16];
  float* out = (float*)d_out;

  const int* row = edges;
  const int* col = edges + N_EDGES;

  char* ws = (char*)d_ws;
  size_t off = 0;
  auto alloc = [&](size_t bytes) -> void* {
    off = (off + 255) & ~(size_t)255;
    void* p = ws + off;
    off += bytes;
    return p;
  };
  __bf16* h0    = (__bf16*)alloc((size_t)N_NODES * HID * 2);
  __bf16* h1    = (__bf16*)alloc((size_t)N_NODES * HID * 2);
  float*  agg   = (float*)alloc((size_t)N_NODES * HID * 4);
  __bf16* pw_e1 = (__bf16*)alloc((size_t)4 * 256 * HID * 2);
  __bf16* pw_e2 = (__bf16*)alloc((size_t)4 * 128 * HID * 2);
  __bf16* pw_n1 = (__bf16*)alloc((size_t)4 * 256 * HID * 2);
  __bf16* pw_n2 = (__bf16*)alloc((size_t)4 * 128 * HID * 2);
  __bf16* pw_d1 = (__bf16*)alloc((size_t)128 * HID * 2);

  // Pack all weights into WMMA B-fragment bf16 layout
  for (int l = 0; l < 4; ++l) {
    pack_b_kernel<<<(256 * 128 + 255) / 256, 256, 0, stream>>>(
        edge_w1 + (size_t)l * 257 * HID, pw_e1 + (size_t)l * 256 * HID, 256, HID);
    pack_b_kernel<<<(128 * 128 + 255) / 256, 256, 0, stream>>>(
        edge_w2 + (size_t)l * 128 * HID, pw_e2 + (size_t)l * 128 * HID, 128, HID);
    pack_b_kernel<<<(256 * 128 + 255) / 256, 256, 0, stream>>>(
        node_w1 + (size_t)l * 256 * HID, pw_n1 + (size_t)l * 256 * HID, 256, HID);
    pack_b_kernel<<<(128 * 128 + 255) / 256, 256, 0, stream>>>(
        node_w2 + (size_t)l * 128 * HID, pw_n2 + (size_t)l * 128 * HID, 128, HID);
  }
  pack_b_kernel<<<(128 * 128 + 255) / 256, 256, 0, stream>>>(dec_w1, pw_d1, 128, HID);

  // Embedding
  {
    long tot = (long)N_NODES * HID;
    embed_kernel<<<(unsigned)((tot + 255) / 256), 256, 0, stream>>>(nodes, emb_w, emb_b, h0);
  }

  __bf16* hc = h0;
  __bf16* hn = h1;
  const int edge_blocks = N_EDGES / (WAVES * 16);              // 10000
  const int node_tiles  = (N_NODES + 15) / 16;                 // 3125
  const int node_blocks = (node_tiles + WAVES - 1) / WAVES;    // 782

  for (int l = 0; l < 4; ++l) {
    long n4 = (long)N_NODES * HID / 4;
    zero_f4_kernel<<<(unsigned)((n4 + 255) / 256), 256, 0, stream>>>((float4*)agg, n4);

    edge_kernel<<<edge_blocks, WAVES * 32, 0, stream>>>(
        hc, row, col, eattr,
        pw_e1 + (size_t)l * 256 * HID,
        edge_w1 + (size_t)l * 257 * HID + (size_t)256 * HID,  // last row (attr col)
        edge_b1 + (size_t)l * HID,
        pw_e2 + (size_t)l * 128 * HID,
        edge_b2 + (size_t)l * HID, agg);

    node_kernel<<<node_blocks, WAVES * 32, 0, stream>>>(
        hc, agg,
        pw_n1 + (size_t)l * 256 * HID, node_b1 + (size_t)l * HID,
        pw_n2 + (size_t)l * 128 * HID, node_b2 + (size_t)l * HID, hn);

    __bf16* t = hc; hc = hn; hn = t;
  }

  dec_kernel<<<node_blocks, WAVES * 32, 0, stream>>>(hc, pw_d1, dec_b1, dec_w2, dec_b2, out);
}
